// LSTM_56959856280387
// MI455X (gfx1250) — matrix operand order repaired and checked by reading, not compiled, not run on hardware
//
#include <hip/hip_runtime.h>
#include <cstdint>
#include <cstddef>

#define T_STEPS 1000
#define NB      32      // batch
#define DIN0    320
#define HID     1024
#define NLAYER  3
#define GATES   4096    // 4*HID
#define NBLK    64      // workgroups in recurrent kernel
#define KMAX    2048    // max K = HID + HID

typedef __attribute__((ext_vector_type(16))) __bf16 v16bf;
typedef __attribute__((ext_vector_type(8)))  float  v8f;

union Frag { v16bf v; uint4 q[2]; };

__device__ __forceinline__ unsigned short f2bf(float f) {
  unsigned u = __float_as_uint(f);
  unsigned r = u + 0x7FFFu + ((u >> 16) & 1u);   // round-to-nearest-even
  return (unsigned short)(r >> 16);
}

__device__ __forceinline__ v8f wmma_bf16(v16bf a, v16bf b, v8f c) {
  // (neg_a, A, neg_b, B, c_mod, C, reuse_a, reuse_b)
  return __builtin_amdgcn_wmma_f32_16x16x32_bf16(false, a, false, b, (short)0, c,
                                                 false, false);
}

// Generic pointers into LDS carry the byte offset in their low 32 bits.
__device__ __forceinline__ unsigned lds_off32(const void* p) {
  return (unsigned)(uintptr_t)p;
}

// CDNA5 async copy global -> LDS, tracked by ASYNCcnt (no VGPR round-trip).
__device__ __forceinline__ void async_load_b128(unsigned lds_byte_off,
                                                const void* gaddr) {
  asm volatile("global_load_async_to_lds_b128 %0, %1, off"
               :: "v"(lds_byte_off), "v"(gaddr) : "memory");
}

__device__ __forceinline__ void wait_asynccnt0() {
  asm volatile("s_wait_asynccnt 0x0" ::: "memory");
}

// Fully-unrolled K-tile GEMM: NKB 32-wide k-blocks, A from global, B from LDS.
// Four independent WMMA chains (even/odd k-block x 2 M-tiles) keep >=3
// instructions between dependent WMMAs (bf16 WMMA RAW hazard window).
template <int NKB>
__device__ __forceinline__ void gemm_ktile(const unsigned short* pa, int strideA,
                                           const unsigned short* pw,
                                           v8f& c0a, v8f& c0b,
                                           v8f& c1a, v8f& c1b) {
#pragma unroll
  for (int i = 0; i < NKB; ++i) {
    const int kb = i * 32;
    Frag a0, a1, b;
    a0.q[0] = *(const uint4*)(pa + kb);
    a0.q[1] = *(const uint4*)(pa + kb + 16);
    a1.q[0] = *(const uint4*)(pa + 16 * strideA + kb);
    a1.q[1] = *(const uint4*)(pa + 16 * strideA + kb + 16);
    b.q[0]  = *(const uint4*)(pw + kb);
    b.q[1]  = *(const uint4*)(pw + kb + 16);
    if (i & 1) {
      c0b = wmma_bf16(a0.v, b.v, c0b);
      c1b = wmma_bf16(a1.v, b.v, c1b);
    } else {
      c0a = wmma_bf16(a0.v, b.v, c0a);
      c1a = wmma_bf16(a1.v, b.v, c1a);
    }
  }
}

// ---------------- prologue kernels ----------------

__global__ void prep_x_kernel(const float* __restrict__ x,
                              unsigned short* __restrict__ xb, int n) {
  int i = blockIdx.x * blockDim.x + threadIdx.x;
  if (i < n) xb[i] = f2bf(x[i]);
}

// Wt[c][k] = (k < Din) ? w_ih[k][c] : w_hh[k - Din][c]   (bf16, row-major (GATES, K))
__global__ void prep_w_kernel(const float* __restrict__ w_ih,
                              const float* __restrict__ w_hh,
                              unsigned short* __restrict__ wt, int Din, int K) {
  long long i = (long long)blockIdx.x * blockDim.x + threadIdx.x;
  long long total = (long long)GATES * K;
  if (i >= total) return;
  int c = (int)(i / K);
  int k = (int)(i % K);
  float v = (k < Din) ? w_ih[(long long)k * GATES + c]
                      : w_hh[(long long)(k - Din) * GATES + c];
  wt[i] = f2bf(v);
}

__global__ void prep_bias_kernel(const float* __restrict__ b_ih,
                                 const float* __restrict__ b_hh,
                                 float* __restrict__ bias, int n) {
  int i = blockIdx.x * blockDim.x + threadIdx.x;
  if (i < n) bias[i] = b_ih[i] + b_hh[i];
}

// reset barrier counter + load h0 into ping buffer (runs before every layer, every call)
__global__ void prep_init_kernel(const float* __restrict__ h0l,
                                 unsigned short* __restrict__ hbuf,
                                 unsigned* __restrict__ cnt) {
  int i = blockIdx.x * blockDim.x + threadIdx.x;
  if (i < NB * HID) hbuf[i] = f2bf(h0l[i]);
  if (i == 0) *cnt = 0u;
}

// ---------------- persistent recurrent kernel (one launch per layer) ----------------
//
// 64 blocks x 256 threads. Block j owns 16 hidden units [j*16, j*16+16):
// its 64 gate columns {g*1024 + j*16 + n}. Wave w -> gate = w&3, ksplit = w>>2.
// Weight slice (64 x K bf16, <=256 KB) is LDS-resident for the whole layer
// (staged once with global_load_async_to_lds_b128). Per step only the
// (32 x K) activation matrix is read from global. DIN is a template constant
// so every GEMM phase has a static trip count and fully unrolls.

template <int DIN>
__global__ __launch_bounds__(256, 1) void lstm_layer_kernel(
    const unsigned short* __restrict__ Ain,  // (T, NB, DIN) bf16
    const unsigned short* __restrict__ wt,   // (GATES, K) bf16, K = DIN + HID
    const float* __restrict__ bias,          // (GATES)
    const float* __restrict__ c0l,           // (NB, HID) fp32
    unsigned short* __restrict__ hbuf,       // (2, NB, HID) bf16 ping-pong
    unsigned short* __restrict__ ys_bf,      // (T, NB, HID) bf16 or null
    float* __restrict__ ys_f32,              // (T, NB, HID) fp32 or null
    float* __restrict__ outHT,               // (NB, HID)
    float* __restrict__ outCT,               // (NB, HID)
    unsigned* __restrict__ cnt)
{
  constexpr int K  = DIN + HID;
  constexpr int Kh = K / 2;                  // 672 or 1024 (32-aligned)

  __shared__ unsigned short swt[64 * KMAX];  // 256 KB weight slice
  __shared__ float sacc[2][NB][64];          // 16 KB split-K gate exchange

  const int j    = blockIdx.x;               // 0..63
  const int tid  = threadIdx.x;
  const int lane = tid & 31;
  const int wv   = tid >> 5;                 // wave 0..7
  const int gate = wv & 3;                   // 0..3 (i,f,g,o)
  const int ks   = wv >> 2;                  // K-split half 0/1
  const int ln   = lane & 15;
  const int hi   = lane >> 4;

  // ---- stage this block's weight slice into LDS (async copy engine) ----
  {
    constexpr int kchunks = K >> 3;          // 16-byte chunks per column row
    constexpr int total   = 64 * kchunks;
    for (int it = tid; it < total; it += 256) {
      int lcol = it / kchunks;
      int koff = (it - lcol * kchunks) * 8;  // in halves
      int g = lcol >> 4, n = lcol & 15;
      int gcol = g * HID + j * 16 + n;
      async_load_b128(lds_off32(&swt[lcol * K + koff]),
                      wt + (size_t)gcol * K + koff);
    }
    wait_asynccnt0();
    __syncthreads();
  }

  // element-wise ownership: thread owns 2 rows of one hidden unit
  const int u    = tid >> 4;                 // 0..15 unit within block
  const int r0   = (tid & 15) * 2;           // row base
  const int gu   = j * 16 + u;               // global hidden index
  const float bI = bias[0 * HID + gu];
  const float bF = bias[1 * HID + gu];
  const float bG = bias[2 * HID + gu];
  const float bO = bias[3 * HID + gu];

  float creg[2];
#pragma unroll
  for (int i = 0; i < 2; ++i) creg[i] = c0l[(r0 + i) * HID + gu];

  const unsigned short* pw = &swt[(gate * 16 + ln) * K + hi * 8];

  for (int t = 0; t < T_STEPS; ++t) {
    const int par = t & 1;
    v8f c0a = {}, c0b = {}, c1a = {}, c1b = {};

    const unsigned short* paX =
        Ain + (size_t)t * NB * DIN + ln * DIN + hi * 8;
    const unsigned short* paH =
        hbuf + par * NB * HID + ln * HID + hi * 8;

    // warm L2 for next step's input slice (global_prefetch)
    if (t + 1 < T_STEPS) {
      __builtin_prefetch(paX + NB * DIN, 0, 1);
      __builtin_prefetch(paX + NB * DIN + 16 * DIN, 0, 1);
    }

    if (ks == 0) {
      // K rows [0, Kh): X part [0, min(DIN,Kh)) then (layer 0 only) H part.
      constexpr int XNKB = (DIN < Kh ? DIN : Kh) / 32;   // 10 or 32
      gemm_ktile<XNKB>(paX, DIN, pw, c0a, c0b, c1a, c1b);
      if constexpr (DIN < Kh) {
        constexpr int HNKB = (Kh - DIN) / 32;            // 11
        gemm_ktile<HNKB>(paH, HID, pw + DIN, c0a, c0b, c1a, c1b);
      }
    } else {
      // K rows [Kh, K): entirely in the H region (Kh >= DIN in both layers).
      constexpr int HNKB = Kh / 32;                      // 21 or 32
      gemm_ktile<HNKB>(paH + (Kh - DIN), HID, pw + Kh, c0a, c0b, c1a, c1b);
    }

    v8f acc0 = c0a + c0b;
    v8f acc1 = c1a + c1b;

    // ---- spill partial gate pre-activations (C layout: VGPR r, lane->row/col) ----
#pragma unroll
    for (int r = 0; r < 8; ++r) {
      int rr = hi * 8 + r;
      sacc[ks][rr]     [gate * 16 + ln] = acc0[r];
      sacc[ks][rr + 16][gate * 16 + ln] = acc1[r];
    }
    __syncthreads();

    // ---- element-wise cell update: reduce split-K, add bias, activate ----
#pragma unroll
    for (int i = 0; i < 2; ++i) {
      int r = r0 + i;
      float ai = sacc[0][r][u]      + sacc[1][r][u]      + bI;
      float af = sacc[0][r][16 + u] + sacc[1][r][16 + u] + bF;
      float ag = sacc[0][r][32 + u] + sacc[1][r][32 + u] + bG;
      float ao = sacc[0][r][48 + u] + sacc[1][r][48 + u] + bO;
      float ig = 1.f / (1.f + __expf(-ai));
      float fg = 1.f / (1.f + __expf(-af));
      float gg = tanhf(ag);
      float og = 1.f / (1.f + __expf(-ao));
      float c  = fg * creg[i] + ig * gg;
      creg[i]  = c;
      float h  = og * tanhf(c);
      hbuf[(par ^ 1) * NB * HID + r * HID + gu] = f2bf(h);
      if (ys_bf)  ys_bf [(size_t)t * NB * HID + r * HID + gu] = f2bf(h);
      if (ys_f32) ys_f32[(size_t)t * NB * HID + r * HID + gu] = h;
      if (t == T_STEPS - 1) {
        outHT[r * HID + gu] = h;
        outCT[r * HID + gu] = c;
      }
    }

    // ---- grid-wide barrier: all 64 blocks must see the new h ----
    __syncthreads();
    if (tid == 0) {
      __threadfence();
      atomicAdd(cnt, 1u);
      unsigned target = (unsigned)(NBLK * (t + 1));
      volatile unsigned* vc = cnt;
      while (*vc < target) { __builtin_amdgcn_s_sleep(8); }
      __threadfence();
    }
    __syncthreads();
  }
}

// ---------------- host launcher ----------------

extern "C" void kernel_launch(void* const* d_in, const int* in_sizes, int n_in,
                              void* d_out, int out_size, void* d_ws, size_t ws_size,
                              hipStream_t stream) {
  (void)in_sizes; (void)n_in; (void)out_size; (void)ws_size;
  const float* x      = (const float*)d_in[0];   // (T, N, D)
  const float* h0     = (const float*)d_in[1];   // (L, N, H)
  const float* c0     = (const float*)d_in[2];   // (L, N, H)
  const float* w_ih0  = (const float*)d_in[3];   // (D, 4H)
  const float* w_rest = (const float*)d_in[4];   // (L-1, H, 4H)
  const float* w_hh   = (const float*)d_in[5];   // (L, H, 4H)
  const float* b_ih   = (const float*)d_in[6];   // (L, 4H)
  const float* b_hh   = (const float*)d_in[7];   // (L, 4H)
  float* out = (float*)d_out;

  char* ws = (char*)d_ws;
  size_t off = 0;
  auto alloc = [&](size_t bytes) -> void* {
    void* p = ws + off;
    off = (off + bytes + 255) & ~(size_t)255;
    return p;
  };
  unsigned short* xb   = (unsigned short*)alloc((size_t)T_STEPS * NB * DIN0 * 2);
  unsigned short* ys0  = (unsigned short*)alloc((size_t)T_STEPS * NB * HID * 2);
  unsigned short* ys1  = (unsigned short*)alloc((size_t)T_STEPS * NB * HID * 2);
  unsigned short* wt0  = (unsigned short*)alloc((size_t)GATES * (DIN0 + HID) * 2);
  unsigned short* wt1  = (unsigned short*)alloc((size_t)GATES * (2 * HID) * 2);
  unsigned short* wt2  = (unsigned short*)alloc((size_t)GATES * (2 * HID) * 2);
  float*          bias = (float*)alloc((size_t)NLAYER * GATES * 4);
  unsigned short* hbuf = (unsigned short*)alloc((size_t)2 * NB * HID * 2);
  unsigned*       cnt  = (unsigned*)alloc(256);

  // prologue: bf16 conversions, fused transposed weights, fused bias
  int nx = T_STEPS * NB * DIN0;
  prep_x_kernel<<<(nx + 255) / 256, 256, 0, stream>>>(x, xb, nx);

  long long tot0 = (long long)GATES * (DIN0 + HID);
  prep_w_kernel<<<(unsigned)((tot0 + 255) / 256), 256, 0, stream>>>(
      w_ih0, w_hh, wt0, DIN0, DIN0 + HID);
  long long tot = (long long)GATES * (2 * HID);
  prep_w_kernel<<<(unsigned)((tot + 255) / 256), 256, 0, stream>>>(
      w_rest, w_hh + (size_t)1 * HID * GATES, wt1, HID, 2 * HID);
  prep_w_kernel<<<(unsigned)((tot + 255) / 256), 256, 0, stream>>>(
      w_rest + (size_t)1 * HID * GATES, w_hh + (size_t)2 * HID * GATES, wt2, HID, 2 * HID);

  int nb = NLAYER * GATES;
  prep_bias_kernel<<<(nb + 255) / 256, 256, 0, stream>>>(b_ih, b_hh, bias, nb);

  float* outHT = out + (size_t)T_STEPS * NB * HID;
  float* outCT = outHT + (size_t)NLAYER * NB * HID;

  // layer 0 (DIN = 320)
  prep_init_kernel<<<(NB * HID + 255) / 256, 256, 0, stream>>>(h0, hbuf, cnt);
  lstm_layer_kernel<DIN0><<<NBLK, 256, 0, stream>>>(
      xb, wt0, bias, c0, hbuf, ys0, nullptr, outHT, outCT, cnt);

  // layer 1 (DIN = 1024)
  prep_init_kernel<<<(NB * HID + 255) / 256, 256, 0, stream>>>(
      h0 + (size_t)1 * NB * HID, hbuf, cnt);
  lstm_layer_kernel<HID><<<NBLK, 256, 0, stream>>>(
      ys0, wt1, bias + (size_t)1 * GATES, c0 + (size_t)1 * NB * HID, hbuf,
      ys1, nullptr, outHT + (size_t)1 * NB * HID, outCT + (size_t)1 * NB * HID, cnt);

  // layer 2 (DIN = 1024) -> fp32 ys straight into d_out
  prep_init_kernel<<<(NB * HID + 255) / 256, 256, 0, stream>>>(
      h0 + (size_t)2 * NB * HID, hbuf, cnt);
  lstm_layer_kernel<HID><<<NBLK, 256, 0, stream>>>(
      ys1, wt2, bias + (size_t)2 * GATES, c0 + (size_t)2 * NB * HID, hbuf,
      nullptr, out, outHT + (size_t)2 * NB * HID, outCT + (size_t)2 * NB * HID, cnt);
}
